// MultiScaleDCN_56100862820884
// MI455X (gfx1250) — matrix operand order repaired and checked by reading, not compiled, not run on hardware
//
#include <hip/hip_runtime.h>
#include <hip/hip_bf16.h>
#include <stdint.h>

// Problem constants (match reference)
#define B_   4
#define H_   64
#define W_   64
#define CIN_ 256
#define G_   8
#define C_   32
#define K_   9
#define NPIX (B_*H_*W_)        // 16384
#define MAXSC 6.0f

typedef __attribute__((ext_vector_type(16))) __bf16 v16bf;
typedef __attribute__((ext_vector_type(8)))  float  v8f;

union Frag { uint4 q[2]; v16bf v; };

__device__ __forceinline__ uint16_t f2bf(float f) {
    uint32_t u = __float_as_uint(f);
    u += 0x7FFFu + ((u >> 16) & 1u);      // round-to-nearest-even
    return (uint16_t)(u >> 16);
}
__device__ __forceinline__ float bf2f(uint16_t h) {
    return __uint_as_float(((uint32_t)h) << 16);
}

// ---------------------------------------------------------------------------
// Elementwise fp32 -> bf16
__global__ __launch_bounds__(256)
void cvt_f32_bf16(const float* __restrict__ in, uint16_t* __restrict__ out, int n) {
    int i = blockIdx.x * 256 + threadIdx.x;
    if (i < n) out[i] = f2bf(in[i]);
}

// Weight convert + transpose: Win [256, Mout] (in-major) -> Wt [MoutPad, 256] bf16,
// zero-padded rows so B-fragment loads never read garbage columns.
__global__ __launch_bounds__(256)
void transpose_w_bf16(const float* __restrict__ Win, uint16_t* __restrict__ Wt,
                      int Mout, int MoutPad) {
    int idx = blockIdx.x * 256 + threadIdx.x;
    int total = MoutPad * CIN_;
    if (idx >= total) return;
    int m = idx / CIN_;
    int k = idx - m * CIN_;
    Wt[idx] = (m < Mout) ? f2bf(Win[(size_t)k * Mout + m]) : (uint16_t)0;
}

// ---------------------------------------------------------------------------
// WMMA GEMM with 2x2 register blocking: each wave computes a 32x32 output tile.
// C[N,M] = A[N,256](bf16) @ Bt[Mpad,256](bf16, out-major) + bias.
// K=256 in 8 steps; 4 x v_wmma_f32_16x16x32_bf16 per step (32 per wave).
// A-frag (16-bit A 16x32): lane<16 -> row=lane, K {0..7 | 16..23};
//                          lane>=16 -> row=lane-16, K {8..15 | 24..31}.
// B-frag (16-bit B 32x16): lane<16 -> col=lane, K {0..7 | 16..23}; lane>=16 -> col-16, K {8..15 | 24..31}.
// C/D: VGPR i, lanes<16 -> M=i, lanes>=16 -> M=i+8; N = lane&15.
template<bool OUT_BF16>
__global__ __launch_bounds__(256)
void wmma_gemm_bf16(const uint16_t* __restrict__ A,
                    const uint16_t* __restrict__ Bt,
                    const float*    __restrict__ bias,
                    void*           __restrict__ Cout,
                    int M, int tilesN) {
    const int lane = threadIdx.x & 31;
    const int wid  = (blockIdx.x * 256 + threadIdx.x) >> 5;
    const int tm   = (wid / tilesN) << 5;   // 32-row tile
    const int tn   = (wid % tilesN) << 5;   // 32-col tile

    const int r  = lane & 15;
    const int hi = (lane >> 4) & 1;

    const uint16_t* arow0 = A  + (size_t)(tm + r) * CIN_ + (hi ? 8  : 0);
    const uint16_t* arow1 = arow0 + (size_t)16 * CIN_;
    const uint16_t* brow0 = Bt + (size_t)(tn + r) * CIN_ + (hi ? 8  : 0);
    const uint16_t* brow1 = brow0 + (size_t)16 * CIN_;

    v8f acc00 = {}, acc01 = {}, acc10 = {}, acc11 = {};
#pragma unroll
    for (int k0 = 0; k0 < CIN_; k0 += 32) {
        Frag a0, a1, b0, b1;
        a0.q[0] = *(const uint4*)(arow0 + k0);
        a0.q[1] = *(const uint4*)(arow0 + k0 + 16);
        a1.q[0] = *(const uint4*)(arow1 + k0);
        a1.q[1] = *(const uint4*)(arow1 + k0 + 16);
        b0.q[0] = *(const uint4*)(brow0 + k0);
        b0.q[1] = *(const uint4*)(brow0 + k0 + 16);
        b1.q[0] = *(const uint4*)(brow1 + k0);
        b1.q[1] = *(const uint4*)(brow1 + k0 + 16);
        acc00 = __builtin_amdgcn_wmma_f32_16x16x32_bf16(false, a0.v, false, b0.v,
                                                        (short)0, acc00, false, false);
        acc01 = __builtin_amdgcn_wmma_f32_16x16x32_bf16(false, a0.v, false, b1.v,
                                                        (short)0, acc01, false, false);
        acc10 = __builtin_amdgcn_wmma_f32_16x16x32_bf16(false, a1.v, false, b0.v,
                                                        (short)0, acc10, false, false);
        acc11 = __builtin_amdgcn_wmma_f32_16x16x32_bf16(false, a1.v, false, b1.v,
                                                        (short)0, acc11, false, false);
    }

    const int col0  = tn + r;
    const int col1  = tn + 16 + r;
    const int rb0   = tm + (hi ? 8 : 0);
    const int rb1   = tm + 16 + (hi ? 8 : 0);

    auto store_tile = [&](const v8f& acc, int rbase, int col) {
        if (col < M) {
            const float bv = bias ? bias[col] : 0.f;
#pragma unroll
            for (int i = 0; i < 8; ++i) {
                float val = acc[i] + bv;
                size_t idx = (size_t)(rbase + i) * M + col;
                if (OUT_BF16) ((uint16_t*)Cout)[idx] = f2bf(val);
                else          ((float*)Cout)[idx]    = val;
            }
        }
    };
    store_tile(acc00, rb0, col0);
    store_tile(acc01, rb0, col1);
    store_tile(acc10, rb1, col0);
    store_tile(acc11, rb1, col1);
}

// ---------------------------------------------------------------------------
// Deformable sampling: one block per pixel, one wave per group, lane = channel.
// Gathers from bf16 v (coalesced 64B per wave), fp32 math, bf16 result.
__global__ __launch_bounds__(256)
void dcn_sample_kernel(const uint16_t* __restrict__ Vb,     // [NPIX, G*C] bf16
                       const float*    __restrict__ QD,     // [NPIX, G*K*2]
                       const float*    __restrict__ QS,     // [NPIX, G*K]
                       const float*    __restrict__ QW,     // [NPIX, G*K]
                       const float*    __restrict__ prior,  // [K*2] (x,y)
                       const float*    __restrict__ dscale, // [G]
                       uint16_t*       __restrict__ Out) {  // [NPIX, G*C] bf16
    const int pix = blockIdx.x;
    const int g   = threadIdx.x >> 5;
    const int c   = threadIdx.x & 31;
    const int b   = pix >> 12;            // / (H*W)
    const int hw  = pix & (H_ * W_ - 1);
    const int h   = hw >> 6;              // / W
    const int w   = hw & (W_ - 1);

    const float  ds   = dscale[g];
    const float* qd   = QD + (size_t)pix * (G_ * K_ * 2) + g * (K_ * 2);
    const float* qs   = QS + (size_t)pix * (G_ * K_)     + g * K_;
    const float* qw   = QW + (size_t)pix * (G_ * K_)     + g * K_;
    const int    chan = g * C_ + c;

    float acc = 0.f;
#pragma unroll
    for (int k = 0; k < K_; ++k) {
        const float sig = 1.f / (1.f + __expf(-(qs[k] + ds)));
        const float m   = sig * MAXSC;
        const float px  = (qd[2 * k + 0] + prior[2 * k + 0]) * m + (float)w;
        const float py  = (qd[2 * k + 1] + prior[2 * k + 1]) * m + (float)h;
        const float wk  = qw[k];

        const float x0f = floorf(px), y0f = floorf(py);
        const int   x0  = (int)x0f,   y0  = (int)y0f;
        const float wx1 = px - x0f, wx0 = 1.f - wx1;
        const float wy1 = py - y0f, wy0 = 1.f - wy1;

        float s00 = 0.f, s01 = 0.f, s10 = 0.f, s11 = 0.f;
        const size_t base = ((size_t)b * H_) * (size_t)W_ * (G_ * C_);
        if ((unsigned)y0 < (unsigned)H_) {
            const size_t rb = base + ((size_t)y0 * W_) * (G_ * C_);
            if ((unsigned)x0       < (unsigned)W_) s00 = bf2f(Vb[rb + (size_t)x0 * (G_ * C_) + chan]);
            if ((unsigned)(x0 + 1) < (unsigned)W_) s01 = bf2f(Vb[rb + (size_t)(x0 + 1) * (G_ * C_) + chan]);
        }
        if ((unsigned)(y0 + 1) < (unsigned)H_) {
            const size_t rb = base + ((size_t)(y0 + 1) * W_) * (G_ * C_);
            if ((unsigned)x0       < (unsigned)W_) s10 = bf2f(Vb[rb + (size_t)x0 * (G_ * C_) + chan]);
            if ((unsigned)(x0 + 1) < (unsigned)W_) s11 = bf2f(Vb[rb + (size_t)(x0 + 1) * (G_ * C_) + chan]);
        }
        acc += wk * (wy0 * (wx0 * s00 + wx1 * s01) + wy1 * (wx0 * s10 + wx1 * s11));
    }
    Out[(size_t)pix * (G_ * C_) + chan] = f2bf(acc);
}

// ---------------------------------------------------------------------------
extern "C" void kernel_launch(void* const* d_in, const int* in_sizes, int n_in,
                              void* d_out, int out_size, void* d_ws, size_t ws_size,
                              hipStream_t stream) {
    (void)in_sizes; (void)n_in; (void)out_size; (void)ws_size;

    const float* x      = (const float*)d_in[0];
    const float* v_w    = (const float*)d_in[1];
    const float* v_b    = (const float*)d_in[2];
    const float* qd_w   = (const float*)d_in[3];
    const float* qd_b   = (const float*)d_in[4];
    const float* qs_w   = (const float*)d_in[5];
    const float* qw_w   = (const float*)d_in[6];
    const float* qw_b   = (const float*)d_in[7];
    const float* out_w  = (const float*)d_in[8];
    const float* out_b  = (const float*)d_in[9];
    const float* prior  = (const float*)d_in[10];
    const float* dscale = (const float*)d_in[11];

    // Workspace carve-up (~44 MB total, 256B aligned)
    char*  ws  = (char*)d_ws;
    size_t off = 0;
    auto carve = [&](size_t bytes) -> char* {
        char* p = ws + off;
        off = (off + bytes + 255) & ~(size_t)255;
        return p;
    };
    uint16_t* Xb    = (uint16_t*)carve((size_t)NPIX * CIN_ * 2);
    uint16_t* WtV   = (uint16_t*)carve((size_t)256 * 256 * 2);
    uint16_t* WtQD  = (uint16_t*)carve((size_t)160 * 256 * 2);
    uint16_t* WtQS  = (uint16_t*)carve((size_t)96  * 256 * 2);
    uint16_t* WtQW  = (uint16_t*)carve((size_t)96  * 256 * 2);
    uint16_t* WtOut = (uint16_t*)carve((size_t)256 * 256 * 2);
    uint16_t* Vb    = (uint16_t*)carve((size_t)NPIX * 256 * 2);
    float*    QD    = (float*)   carve((size_t)NPIX * 144 * 4);
    float*    QS    = (float*)   carve((size_t)NPIX * 72  * 4);
    float*    QW    = (float*)   carve((size_t)NPIX * 72  * 4);
    uint16_t* Dcn   = (uint16_t*)carve((size_t)NPIX * 256 * 2);

    // Stage 0: precision staging (x once; weights are tiny)
    cvt_f32_bf16<<<(NPIX * CIN_ + 255) / 256, 256, 0, stream>>>(x, Xb, NPIX * CIN_);
    transpose_w_bf16<<<(256 * 256 + 255) / 256, 256, 0, stream>>>(v_w,  WtV,  256, 256);
    transpose_w_bf16<<<(160 * 256 + 255) / 256, 256, 0, stream>>>(qd_w, WtQD, 144, 160);
    transpose_w_bf16<<<(96  * 256 + 255) / 256, 256, 0, stream>>>(qs_w, WtQS, 72,  96);
    transpose_w_bf16<<<(96  * 256 + 255) / 256, 256, 0, stream>>>(qw_w, WtQW, 72,  96);
    transpose_w_bf16<<<(256 * 256 + 255) / 256, 256, 0, stream>>>(out_w, WtOut, 256, 256);

    // Stage 1: projection GEMMs (shared bf16 LHS), 8 waves / block, 32x32 per wave
    const int rowTiles32 = NPIX / 32;  // 512
    wmma_gemm_bf16<true ><<<rowTiles32 * 8 / 8, 256, 0, stream>>>(Xb, WtV,  v_b,     Vb,  256, 8);
    wmma_gemm_bf16<false><<<rowTiles32 * 5 / 8, 256, 0, stream>>>(Xb, WtQD, qd_b,    QD,  144, 5);
    wmma_gemm_bf16<false><<<rowTiles32 * 3 / 8, 256, 0, stream>>>(Xb, WtQS, nullptr, QS,  72,  3);
    wmma_gemm_bf16<false><<<rowTiles32 * 3 / 8, 256, 0, stream>>>(Xb, WtQW, qw_b,    QW,  72,  3);

    // Stage 2: deformable bilinear sampling
    dcn_sample_kernel<<<NPIX, 256, 0, stream>>>(Vb, QD, QS, QW, prior, dscale, Dcn);

    // Stage 3: output projection -> fp32 d_out
    wmma_gemm_bf16<false><<<rowTiles32 * 8 / 8, 256, 0, stream>>>(Dcn, WtOut, out_b,
                                                                  (float*)d_out, 256, 8);
}
